// VoxelFeatureExtractor_24524263260766
// MI455X (gfx1250) — matrix operand order repaired and checked by reading, not compiled, not run on hardware
//
#include <hip/hip_runtime.h>
#include <hip/hip_bf16.h>

typedef __attribute__((ext_vector_type(2)))  float    v2f;
typedef __attribute__((ext_vector_type(8)))  float    v8f;
typedef __attribute__((ext_vector_type(8)))  _Float16 v8h;
typedef __attribute__((ext_vector_type(16))) _Float16 v16h;

#define P_   32
#define C_   4
#define H_   64
#define O_   128
#define EPS_ 1e-5f
#define STRIDE 72   // halfwords per staged row: 64 data + 8 pad (16B-aligned rows, staggered banks)

// single-instruction ReLU: v_med3_f32(x, 0, +inf)
__device__ __forceinline__ float relu1(float x) {
  return __builtin_amdgcn_fmed3f(x, 0.0f, __builtin_inff());
}

__global__ __launch_bounds__(256) void pfn_fused_kernel(
    const float* __restrict__ vf, const long long* __restrict__ vnp,
    const float* __restrict__ W1, const float* __restrict__ b1,
    const float* __restrict__ g1, const float* __restrict__ be1,
    const float* __restrict__ m1, const float* __restrict__ v1,
    const float* __restrict__ W2, const float* __restrict__ b2,
    const float* __restrict__ g2, const float* __restrict__ be2,
    const float* __restrict__ m2, const float* __restrict__ v2,
    const float* __restrict__ W3, const float* __restrict__ b3,
    float* __restrict__ out, int nv)
{
  __shared__ alignas(16) _Float16 w3t[O_ * STRIDE];       // (W3)^T as f16: [n][k]
  __shared__ alignas(16) _Float16 w2t[H_ * STRIDE];       // (W2 * s2)^T as f16: [n][k]
  __shared__ alignas(16) _Float16 stag[8][16 * STRIDE];   // per-wave 16x64 transpose staging

  const int tid  = threadIdx.x;
  const int lane = tid & 31;
  const int wid  = tid >> 5;
  const int g    = lane >> 4;   // half-wave group (0/1)
  const int ln   = lane & 15;

  // ---- preamble: build W3^T and BN2-folded W2^T (f16) in LDS, cooperatively ----
  for (int i = tid; i < O_ * H_; i += 256) {
    int n = i >> 6;   // 0..127
    int k = i & 63;   // 0..63
    w3t[n * STRIDE + k] = (_Float16)W3[k * O_ + n];
  }
  for (int i = tid; i < H_ * H_; i += 256) {
    int n = i >> 6;   // 0..63
    int k = i & 63;   // 0..63
    float s2 = g2[n] * rsqrtf(v2[n] + EPS_);
    w2t[n * STRIDE + k] = (_Float16)(W2[k * H_ + n] * s2);
  }
  __syncthreads();

  // ---- per-lane folded constants + BN1-folded layer-1 B fragments (f32 4x16) ----
  v2f bw1[4];
  float b1c[4], b2c[4], b3c[8];
#pragma unroll
  for (int t = 0; t < 4; ++t) {
    int n = ln + 16 * t;
    float s1 = g1[n] * rsqrtf(v1[n] + EPS_);
    bw1[t].x = W1[(2 * g) * H_ + n] * s1;       // B frag 4x16: vgpr i holds k = 2g+i
    bw1[t].y = W1[(2 * g + 1) * H_ + n] * s1;
    b1c[t] = (b1[n] - m1[n]) * s1 + be1[n];
    float s2 = g2[n] * rsqrtf(v2[n] + EPS_);
    b2c[t] = (b2[n] - m2[n]) * s2 + be2[n];
  }
#pragma unroll
  for (int t = 0; t < 8; ++t) b3c[t] = b3[ln + 16 * t];

  _Float16* st = &stag[wid][0];
  // single-base addressing so ds ops use one VGPR + immediate offsets
  _Float16*       sb  = st  + (8 * g) * STRIDE + ln;   // C-layout scatter base
  const _Float16* ab  = st  + ln * STRIDE + 8 * g;     // A-fragment gather base
  const _Float16* b2b = w2t + ln * STRIDE + 8 * g;     // W2 B-fragment base
  const _Float16* b3b = w3t + ln * STRIDE + 8 * g;     // W3 B-fragment base

  const int totalWaves = (int)((gridDim.x * blockDim.x) >> 5);
  int v = (int)((blockIdx.x * blockDim.x + tid) >> 5);

  for (; v < nv; v += totalWaves) {
    if (v + totalWaves < nv)
      __builtin_prefetch(vf + (size_t)(v + totalWaves) * (P_ * C_), 0, 0);

    const int np = (int)vnp[v];              // wave-uniform
    const int nhalves = (np > 16) ? 2 : 1;   // skip fully-masked second half (uniform)

    float mxt[8];
#pragma unroll
    for (int t = 0; t < 8; ++t) mxt[t] = -__builtin_inff();

    const float* xbase = vf + (size_t)v * (P_ * C_) + ln * C_ + 2 * g;

    for (int half = 0; half < nhalves; ++half) {
      // additive row-validity mask for this half: 0 if row valid, -inf otherwise
      float vmask[8];
#pragma unroll
      for (int r = 0; r < 8; ++r)
        vmask[r] = (half * 16 + r + 8 * g < np) ? 0.0f : -__builtin_inff();

      // ======== layer 1: X(16x4) @ (W1*s1)(4x64) + b1c, fp32 WMMA ========
      {
        const float* xrow = xbase + half * (16 * C_);
        v2f ax; ax.x = xrow[0]; ax.y = xrow[1];   // A frag 16x4: vgpr i holds k = 2g+i
#pragma unroll
        for (int t = 0; t < 4; ++t) {
          v8f c;
#pragma unroll
          for (int r = 0; r < 8; ++r) c[r] = b1c[t];   // folded bias in accumulator
          c = __builtin_amdgcn_wmma_f32_16x16x4_f32(false, ax, false, bw1[t],
                                                    (short)0, c, false, false);
#pragma unroll
          for (int r = 0; r < 8; ++r)
            sb[r * STRIDE + 16 * t] = (_Float16)relu1(c[r]);
        }
      }

      // ======== layer 2: H1(16x64) @ (W2*s2)(64x64) + b2c, f16 WMMA ========
      v16h af[2];
#pragma unroll
      for (int j = 0; j < 2; ++j) {
        v8h lo = *(const v8h*)(ab + 32 * j);
        v8h hi = *(const v8h*)(ab + 32 * j + 16);
        af[j] = __builtin_shufflevector(lo, hi, 0,1,2,3,4,5,6,7,8,9,10,11,12,13,14,15);
      }
#pragma unroll
      for (int t = 0; t < 4; ++t) {
        v8f c;
#pragma unroll
        for (int r = 0; r < 8; ++r) c[r] = b2c[t];
#pragma unroll
        for (int j = 0; j < 2; ++j) {
          v8h lo = *(const v8h*)(b2b + t * (16 * STRIDE) + 32 * j);
          v8h hi = *(const v8h*)(b2b + t * (16 * STRIDE) + 32 * j + 16);
          v16h bf = __builtin_shufflevector(lo, hi, 0,1,2,3,4,5,6,7,8,9,10,11,12,13,14,15);
          c = __builtin_amdgcn_wmma_f32_16x16x32_f16(false, af[j], false, bf,
                                                     (short)0, c, false, false);
        }
#pragma unroll
        for (int r = 0; r < 8; ++r)
          sb[r * STRIDE + 16 * t] = (_Float16)relu1(c[r]);
      }

      // ======== layer 3: H2(16x64) @ W3(64x128) + b3, f16 WMMA ========
      v16h af3[2];
#pragma unroll
      for (int j = 0; j < 2; ++j) {
        v8h lo = *(const v8h*)(ab + 32 * j);
        v8h hi = *(const v8h*)(ab + 32 * j + 16);
        af3[j] = __builtin_shufflevector(lo, hi, 0,1,2,3,4,5,6,7,8,9,10,11,12,13,14,15);
      }
#pragma unroll
      for (int t = 0; t < 8; ++t) {
        v8f c;
#pragma unroll
        for (int r = 0; r < 8; ++r) c[r] = b3c[t];
#pragma unroll
        for (int j = 0; j < 2; ++j) {
          v8h lo = *(const v8h*)(b3b + t * (16 * STRIDE) + 32 * j);
          v8h hi = *(const v8h*)(b3b + t * (16 * STRIDE) + 32 * j + 16);
          v16h bf = __builtin_shufflevector(lo, hi, 0,1,2,3,4,5,6,7,8,9,10,11,12,13,14,15);
          c = __builtin_amdgcn_wmma_f32_16x16x32_f16(false, af3[j], false, bf,
                                                     (short)0, c, false, false);
        }
        // masked running max via additive mask (-inf kills invalid rows)
#pragma unroll
        for (int r = 0; r < 8; ++r)
          mxt[t] = fmaxf(mxt[t], c[r] + vmask[r]);
      }
    }

    // combine the two half-wave row groups (lanes l <-> l+16 hold same n) and store
#pragma unroll
    for (int t = 0; t < 8; ++t) {
      float o = fmaxf(mxt[t], __shfl_xor(mxt[t], 16));
      if (lane < 16) out[(size_t)v * O_ + ln + 16 * t] = o;
    }
  }
}

extern "C" void kernel_launch(void* const* d_in, const int* in_sizes, int n_in,
                              void* d_out, int out_size, void* d_ws, size_t ws_size,
                              hipStream_t stream) {
  (void)n_in; (void)d_ws; (void)ws_size; (void)out_size;
  const float*     vf  = (const float*)d_in[0];
  const long long* vnp = (const long long*)d_in[1];
  const float* W1 = (const float*)d_in[2];
  const float* b1 = (const float*)d_in[3];
  const float* g1 = (const float*)d_in[4];
  const float* be1= (const float*)d_in[5];
  const float* m1 = (const float*)d_in[6];
  const float* v1 = (const float*)d_in[7];
  const float* W2 = (const float*)d_in[8];
  const float* b2 = (const float*)d_in[9];
  const float* g2 = (const float*)d_in[10];
  const float* be2= (const float*)d_in[11];
  const float* m2 = (const float*)d_in[12];
  const float* v2 = (const float*)d_in[13];
  const float* W3 = (const float*)d_in[14];
  const float* b3 = (const float*)d_in[15];
  float* out = (float*)d_out;
  const int nv = in_sizes[1];   // voxel_num_points element count == NV

  pfn_fused_kernel<<<1024, 256, 0, stream>>>(vf, vnp, W1, b1, g1, be1, m1, v1,
                                             W2, b2, g2, be2, m2, v2, W3, b3,
                                             out, nv);
}
